// AVMambaClassifier_40853728920181
// MI455X (gfx1250) — compile-verified, matrix operand order checked by reading, not run on hardware
//
#include <hip/hip_runtime.h>
#include <math.h>

typedef _Float16 half_t;
typedef __attribute__((ext_vector_type(8)))  _Float16 v8h;
typedef __attribute__((ext_vector_type(16))) _Float16 v16h;
typedef __attribute__((ext_vector_type(8)))  float    v8f;
typedef __attribute__((ext_vector_type(4)))  int      v4i;

#define B_   32
#define T_   512
#define D_   2048
#define DM_  256
#define L_   4
#define DI_  512
#define DS_  16
#define DC_  4
#define R_   16
#define H_   128
#define NC_  8
#define MTOK (B_ * T_)   // 16384 tokens

// ---------------------------------------------------------------------------
// Optional CDNA5 async global->LDS path (ASYNCcnt), probed at compile time
// ---------------------------------------------------------------------------
#if defined(__has_builtin)
#if __has_builtin(__builtin_amdgcn_global_load_async_to_lds_b128) && \
    __has_builtin(__builtin_amdgcn_s_wait_asynccnt)
#define USE_ASYNC_LDS 1
#endif
#endif

#ifdef USE_ASYNC_LDS
__device__ __forceinline__ void async_copy16(const half_t* g, half_t* l) {
  // builtin signature (probe-confirmed param1): (int4* src, int4* lds_dst, imm offset, imm cpol)
  __builtin_amdgcn_global_load_async_to_lds_b128((v4i*)g, (v4i*)l, 0, 0);
}
#endif

// ---------------------------------------------------------------------------
// f32 -> f16 conversion
// ---------------------------------------------------------------------------
__global__ __launch_bounds__(256)
void f32_to_f16(const float* __restrict__ src, half_t* __restrict__ dst, int n) {
  int i = blockIdx.x * 256 + threadIdx.x;
  if (i < n) dst[i] = (half_t)src[i];
}

// ---------------------------------------------------------------------------
// Masked temporal instance norm over T per (b, d); writes f16, masked -> 0
// ---------------------------------------------------------------------------
__global__ __launch_bounds__(256)
void inst_norm(const float* __restrict__ x, const int* __restrict__ lengths,
               const float* __restrict__ gamma, const float* __restrict__ beta,
               half_t* __restrict__ xn16) {
  int d = blockIdx.x * 256 + threadIdx.x;
  int b = blockIdx.y;
  int len = lengths[b];
  float denom = (float)(len > 0 ? len : 1);
  const float* xb = x + (size_t)b * T_ * D_ + d;
  float s = 0.f, s2 = 0.f;
  for (int t = 0; t < T_; ++t) {
    if (t < len) { float v = xb[(size_t)t * D_]; s += v; s2 += v * v; }
  }
  float mu  = s / denom;
  float var = s2 / denom - mu * mu;
  float rstd = rsqrtf(var + 1e-5f);
  float g = gamma[d], be = beta[d];
  half_t* ob = xn16 + (size_t)b * T_ * D_ + d;
  for (int t = 0; t < T_; ++t) {
    float v = 0.f;
    if (t < len) v = (xb[(size_t)t * D_] - mu) * rstd * g + be;
    ob[(size_t)t * D_] = (half_t)v;
  }
}

// ---------------------------------------------------------------------------
// LayerNorm over DM_=256 per row; optional f16 and/or f32 outputs
// ---------------------------------------------------------------------------
__global__ __launch_bounds__(256)
void layer_norm(const float* __restrict__ X, const float* __restrict__ g,
                const float* __restrict__ bta, half_t* o16, float* o32) {
  __shared__ float red[256];
  __shared__ float s_mu, s_rs;
  int row = blockIdx.x, tid = threadIdx.x;
  float v = X[(size_t)row * DM_ + tid];
  red[tid] = v; __syncthreads();
  for (int s = 128; s > 0; s >>= 1) { if (tid < s) red[tid] += red[tid + s]; __syncthreads(); }
  if (tid == 0) s_mu = red[0] * (1.f / DM_);
  __syncthreads();
  float dv = v - s_mu;
  red[tid] = dv * dv; __syncthreads();
  for (int s = 128; s > 0; s >>= 1) { if (tid < s) red[tid] += red[tid + s]; __syncthreads(); }
  if (tid == 0) s_rs = rsqrtf(red[0] * (1.f / DM_) + 1e-5f);
  __syncthreads();
  float y = dv * s_rs * g[tid] + bta[tid];
  if (o16) o16[(size_t)row * DM_ + tid] = (half_t)y;
  if (o32) o32[(size_t)row * DM_ + tid] = y;
}

// ---------------------------------------------------------------------------
// WMMA GEMM, K is compile-time (multiple of 32): C[M,N] = A[M,K] * Bw[N,K]^T.
// Block = 8 waves = 128 rows x 64 cols. B tile (64x32 f16 = 4KB) is staged in
// LDS, double-buffered, filled by async global->LDS DMA (ASYNCcnt) when
// available. A fragments are per-wave-private direct b128 loads + prefetch.
// Epilogue modes: 0: acc(+bias)   1: Cin + scale*acc   2: tanh(acc+bias)
// ---------------------------------------------------------------------------
template <int K>
__global__ __launch_bounds__(256)
void gemm_f16_wmma(const half_t* __restrict__ A, const half_t* __restrict__ Bw,
                   float* C, const float* Cin,
                   const float* __restrict__ bias,
                   const float* __restrict__ scale_ptr,
                   int N, int mode) {
  __shared__ half_t sB[2][64 * 32];     // 2 x 4KB B tiles

  const int tid     = threadIdx.x;
  const int lane    = tid & 31;
  const int wave    = tid >> 5;
  const int half_id = lane >> 4;        // K-half selector (ISA A/B layout)
  const int l16     = lane & 15;        // M row (A) / N col (B,C,D)
  const int m0      = blockIdx.x * 128 + wave * 16;
  const int n0      = blockIdx.y * 64;

  // B-tile fill mapping: thread -> (row, 16B chunk). Clamp rows past N.
  const int fr = tid >> 2;              // 0..63
  const int fc = tid & 3;               // 0..3
  int frow = n0 + fr; if (frow >= N) frow = N - 1;
  const half_t* gB = Bw + (size_t)frow * K + fc * 8;
  half_t* lB0 = &sB[0][fr * 32 + fc * 8];
  half_t* lB1 = &sB[1][fr * 32 + fc * 8];

  auto fill = [&](int buf, int k0) {
    half_t* l = buf ? lB1 : lB0;
#ifdef USE_ASYNC_LDS
    async_copy16(gB + k0, l);
#else
    *(v8h*)l = *(const v8h*)(gB + k0);
#endif
  };

  const half_t* pA = A + (size_t)(m0 + l16) * K;
  v8f acc[4];
#pragma unroll
  for (int nt = 0; nt < 4; ++nt)
    acc[nt] = (v8f){0.f, 0.f, 0.f, 0.f, 0.f, 0.f, 0.f, 0.f};

  fill(0, 0);                           // prologue
  int cur = 0;
  for (int k0 = 0; k0 < K; k0 += 32, cur ^= 1) {
    __syncthreads();                    // prior consumers of buf cur^1 done
    if (k0 + 32 < K) fill(cur ^ 1, k0 + 32);
#ifdef USE_ASYNC_LDS
    if (k0 + 32 < K) __builtin_amdgcn_s_wait_asynccnt(1);   // cur tile landed
    else             __builtin_amdgcn_s_wait_asynccnt(0);
#endif
    __syncthreads();                    // cur tile visible to all waves

    const int kA = k0 + half_id * 8;
    v8h alo = *(const v8h*)(pA + kA);
    v8h ahi = *(const v8h*)(pA + kA + 16);
    v16h a = __builtin_shufflevector(alo, ahi,
        0,1,2,3,4,5,6,7,8,9,10,11,12,13,14,15);
    if (k0 + 64 < K) __builtin_prefetch(pA + k0 + 64);      // global_prefetch_b8

    const v8h* sbp = (const v8h*)&sB[cur][0];
#pragma unroll
    for (int nt = 0; nt < 4; ++nt) {
      const int nn = nt * 16 + l16;                 // tile-local B row
      v8h blo = sbp[nn * 4 + half_id];
      v8h bhi = sbp[nn * 4 + 2 + half_id];
      v16h b = __builtin_shufflevector(blo, bhi,
          0,1,2,3,4,5,6,7,8,9,10,11,12,13,14,15);
      acc[nt] = __builtin_amdgcn_wmma_f32_16x16x32_f16(
          false, a, false, b, (short)0, acc[nt], false, false);
    }
  }

  float scale = scale_ptr ? scale_ptr[0] : 1.0f;
#pragma unroll
  for (int nt = 0; nt < 4; ++nt) {
    int n = n0 + nt * 16 + l16;
    if (n >= N) continue;
    float bval = bias ? bias[n] : 0.0f;
#pragma unroll
    for (int r = 0; r < 8; ++r) {
      int m = m0 + r + 8 * half_id;     // D layout: row = r + 8*(lane>>4)
      float v = acc[nt][r];
      size_t idx = (size_t)m * N + n;
      if (mode == 0)      C[idx] = v + bval;
      else if (mode == 1) C[idx] = Cin[idx] + scale * v;
      else                C[idx] = tanhf(v + bval);
    }
  }
}

// ---------------------------------------------------------------------------
// Depthwise causal conv (DC=4) over T + bias + SiLU; reads x-half of xz
// ---------------------------------------------------------------------------
__global__ __launch_bounds__(256)
void conv_silu(const float* __restrict__ xz, const float* __restrict__ cw,
               const float* __restrict__ cb, half_t* __restrict__ xc16) {
  int d = blockIdx.x * 256 + threadIdx.x;   // channel 0..511
  int b = blockIdx.y;
  float w0 = cw[d * DC_ + 0], w1 = cw[d * DC_ + 1];
  float w2 = cw[d * DC_ + 2], w3 = cw[d * DC_ + 3];
  float bias = cb[d];
  float p1 = 0.f, p2 = 0.f, p3 = 0.f;
  const float* xp = xz + (size_t)b * T_ * (2 * DI_) + d;
  half_t* op = xc16 + (size_t)b * T_ * DI_ + d;
  for (int t = 0; t < T_; ++t) {
    float cur = xp[(size_t)t * (2 * DI_)];
    float v = p3 * w0 + p2 * w1 + p1 * w2 + cur * w3 + bias;
    v = v / (1.f + __expf(-v));             // SiLU
    op[(size_t)t * DI_] = (half_t)v;
    p3 = p2; p2 = p1; p1 = cur;
  }
}

// ---------------------------------------------------------------------------
// Selective scan: block = one batch, thread = one channel d of DI_=512.
// Per step t, the 48-float (dtr|B|C) row is staged in LDS and broadcast.
// dt = softplus(dbc[:16]·dtw + dtb) computed inline; SiLU(z) gate fused.
// ---------------------------------------------------------------------------
__global__ __launch_bounds__(512)
void mamba_scan(const float* __restrict__ dbc, const half_t* __restrict__ xc16,
                const float* __restrict__ xz, const float* __restrict__ dtw,
                const float* __restrict__ dtb, const float* __restrict__ A_log,
                const float* __restrict__ Dp, half_t* __restrict__ y16) {
  __shared__ float sd[48];
  int d = threadIdx.x;
  int b = blockIdx.x;
  float Ar[DS_], Wr[R_], hs[DS_];
#pragma unroll
  for (int j = 0; j < DS_; ++j) {
    Ar[j] = -__expf(A_log[(size_t)d * DS_ + j]);
    Wr[j] = dtw[(size_t)d * R_ + j];
    hs[j] = 0.f;
  }
  float bdt = dtb[d], dpar = Dp[d];
  for (int t = 0; t < T_; ++t) {
    __syncthreads();
    if (d < 48) sd[d] = dbc[((size_t)b * T_ + t) * 48 + d];
    __syncthreads();
    float dv = bdt;
#pragma unroll
    for (int j = 0; j < R_; ++j) dv += sd[j] * Wr[j];
    float dt = (dv > 20.f) ? dv : log1pf(__expf(dv));   // softplus
    float xt = (float)xc16[((size_t)b * T_ + t) * DI_ + d];
    float y = 0.f;
#pragma unroll
    for (int j = 0; j < DS_; ++j) {
      hs[j] = __expf(dt * Ar[j]) * hs[j] + (dt * xt) * sd[16 + j];
      y += hs[j] * sd[32 + j];
    }
    y += dpar * xt;
    float zv = xz[((size_t)b * T_ + t) * (2 * DI_) + DI_ + d];
    float out = y * (zv / (1.f + __expf(-zv)));
    y16[((size_t)b * T_ + t) * DI_ + d] = (half_t)out;
  }
}

// ---------------------------------------------------------------------------
// s[m] = t1[m,:] . attn_w2
// ---------------------------------------------------------------------------
__global__ __launch_bounds__(128)
void attn_score(const float* __restrict__ t1, const float* __restrict__ w2,
                float* __restrict__ s) {
  __shared__ float red[128];
  int m = blockIdx.x, tid = threadIdx.x;
  red[tid] = t1[(size_t)m * H_ + tid] * w2[tid];
  __syncthreads();
  for (int st = 64; st > 0; st >>= 1) { if (tid < st) red[tid] += red[tid + st]; __syncthreads(); }
  if (tid == 0) s[m] = red[0];
}

// ---------------------------------------------------------------------------
// Masked softmax over T + weighted context sum; block = one batch
// ---------------------------------------------------------------------------
__global__ __launch_bounds__(256)
void softmax_ctx(const float* __restrict__ s, const float* __restrict__ hn,
                 const int* __restrict__ lengths, float* __restrict__ ctx) {
  __shared__ float sv[T_];
  __shared__ float red[256];
  __shared__ float s_max, s_den;
  int b = blockIdx.x, tid = threadIdx.x;
  int len = lengths[b];
  for (int t = tid; t < T_; t += 256)
    sv[t] = (t < len) ? s[(size_t)b * T_ + t] : -INFINITY;
  __syncthreads();
  red[tid] = fmaxf(sv[tid], sv[tid + 256]);
  __syncthreads();
  for (int st = 128; st > 0; st >>= 1) { if (tid < st) red[tid] = fmaxf(red[tid], red[tid + st]); __syncthreads(); }
  if (tid == 0) s_max = red[0];
  __syncthreads();
  float sum0 = 0.f;
  for (int t = tid; t < T_; t += 256) {
    float e = (t < len) ? __expf(sv[t] - s_max) : 0.f;
    sv[t] = e; sum0 += e;
  }
  red[tid] = sum0; __syncthreads();
  for (int st = 128; st > 0; st >>= 1) { if (tid < st) red[tid] += red[tid + st]; __syncthreads(); }
  if (tid == 0) s_den = 1.f / fmaxf(red[0], 1e-9f);
  __syncthreads();
  float acc = 0.f;
  const float* hb = hn + (size_t)b * T_ * DM_ + tid;   // thread owns dim tid
  for (int t = 0; t < T_; ++t) acc += sv[t] * hb[(size_t)t * DM_];
  ctx[(size_t)b * DM_ + tid] = acc * s_den;
}

// ---------------------------------------------------------------------------
// out[b,c] = ctx[b,:] . head_w[c,:] + head_b[c]
// ---------------------------------------------------------------------------
__global__ __launch_bounds__(256)
void head_kernel(const float* __restrict__ ctx, const float* __restrict__ hw,
                 const float* __restrict__ hb, float* __restrict__ out) {
  int tid = threadIdx.x;               // 256 = 32 batches * 8 classes
  int b = tid >> 3, c = tid & 7;
  float acc = hb[c];
  const float* cb = ctx + (size_t)b * DM_;
  const float* wr = hw + (size_t)c * DM_;
  for (int k = 0; k < DM_; ++k) acc += cb[k] * wr[k];
  out[tid] = acc;
}

// ---------------------------------------------------------------------------
extern "C" void kernel_launch(void* const* d_in, const int* in_sizes, int n_in,
                              void* d_out, int out_size, void* d_ws, size_t ws_size,
                              hipStream_t stream) {
  (void)in_sizes; (void)n_in; (void)out_size; (void)ws_size;
  const float* x       = (const float*)d_in[0];
  const int*   lengths = (const int*)  d_in[1];
  const float* tin_g   = (const float*)d_in[2];
  const float* tin_b   = (const float*)d_in[3];
  const float* proj_w  = (const float*)d_in[4];
  const float* proj_b  = (const float*)d_in[5];
  const float* ln_g    = (const float*)d_in[6];
  const float* ln_b    = (const float*)d_in[7];
  const float* in_w    = (const float*)d_in[8];
  const float* conv_w  = (const float*)d_in[9];
  const float* conv_b  = (const float*)d_in[10];
  const float* x_w     = (const float*)d_in[11];
  const float* dt_w    = (const float*)d_in[12];
  const float* dt_b    = (const float*)d_in[13];
  const float* A_log   = (const float*)d_in[14];
  const float* D_par   = (const float*)d_in[15];
  const float* out_w   = (const float*)d_in[16];
  const float* oln_g   = (const float*)d_in[17];
  const float* oln_b   = (const float*)d_in[18];
  const float* res_s   = (const float*)d_in[19];
  const float* a_w1    = (const float*)d_in[20];
  const float* a_b1    = (const float*)d_in[21];
  const float* a_w2    = (const float*)d_in[22];
  const float* head_w  = (const float*)d_in[23];
  const float* head_b  = (const float*)d_in[24];
  float* out = (float*)d_out;

  // ---- workspace bump allocator (lifetime-based aliasing) ----
  char* ws = (char*)d_ws;
  size_t off = 0;
  auto alloc = [&](size_t bytes) -> char* {
    char* p = ws + off;
    off += (bytes + 255) & ~(size_t)255;
    return p;
  };
  half_t* w16_proj = (half_t*)alloc((size_t)DM_ * D_ * 2);
  half_t* w16_in   = (half_t*)alloc((size_t)L_ * 2 * DI_ * DM_ * 2);
  half_t* w16_x    = (half_t*)alloc((size_t)L_ * 48 * DI_ * 2);
  half_t* w16_out  = (half_t*)alloc((size_t)L_ * DM_ * DI_ * 2);
  half_t* w16_a1   = (half_t*)alloc((size_t)H_ * DM_ * 2);
  float*  h        = (float*) alloc((size_t)MTOK * DM_ * 4);
  char*   ln16buf  = alloc((size_t)MTOK * DM_ * 2);   // ln16 -> hn16
  char*   bigbuf   = alloc((size_t)MTOK * D_ * 2);    // xn16 -> xz (64 MB)
  char*   xcbuf    = alloc((size_t)MTOK * DI_ * 2);   // xc16 -> t1
  char*   dbcbuf   = alloc((size_t)MTOK * 48 * 4);    // dbc  -> s, ctx
  char*   ybuf     = alloc((size_t)MTOK * DI_ * 2);   // y16  -> hn (f32, same size)

  half_t* xn16 = (half_t*)bigbuf;
  float*  xz   = (float*) bigbuf;   // MTOK*1024*4 == MTOK*2048*2
  half_t* ln16 = (half_t*)ln16buf;
  half_t* hn16 = (half_t*)ln16buf;
  half_t* xc16 = (half_t*)xcbuf;
  float*  t1   = (float*) xcbuf;    // MTOK*128*4 <= MTOK*512*2
  float*  dbc  = (float*) dbcbuf;
  float*  sbuf = (float*) dbcbuf;   // 64 KB, dbc dead by then
  float*  ctxb = (float*) (dbcbuf + (1 << 20));
  half_t* y16  = (half_t*)ybuf;
  float*  hn   = (float*) ybuf;     // MTOK*256*4 == MTOK*512*2

  // ---- convert weights to f16 (once per call; deterministic) ----
  auto cvt = [&](const float* s_, half_t* d_, int n) {
    f32_to_f16<<<(n + 255) / 256, 256, 0, stream>>>(s_, d_, n);
  };
  cvt(proj_w, w16_proj, DM_ * D_);
  cvt(in_w,   w16_in,   L_ * 2 * DI_ * DM_);
  cvt(x_w,    w16_x,    L_ * 48 * DI_);
  cvt(out_w,  w16_out,  L_ * DM_ * DI_);
  cvt(a_w1,   w16_a1,   H_ * DM_);

  // ---- masked temporal instance norm ----
  inst_norm<<<dim3(D_ / 256, B_), 256, 0, stream>>>(x, lengths, tin_g, tin_b, xn16);

  // ---- input projection: h = xn @ proj_w^T + proj_b ----
  gemm_f16_wmma<D_><<<dim3(MTOK / 128, DM_ / 64), 256, 0, stream>>>(
      xn16, w16_proj, h, nullptr, proj_b, nullptr, DM_, 0);

  // ---- Mamba layers ----
  for (int l = 0; l < L_; ++l) {
    layer_norm<<<MTOK, 256, 0, stream>>>(h, ln_g + l * DM_, ln_b + l * DM_, ln16, nullptr);
    gemm_f16_wmma<DM_><<<dim3(MTOK / 128, (2 * DI_) / 64), 256, 0, stream>>>(
        ln16, w16_in + (size_t)l * 2 * DI_ * DM_, xz, nullptr, nullptr, nullptr,
        2 * DI_, 0);
    conv_silu<<<dim3(DI_ / 256, B_), 256, 0, stream>>>(
        xz, conv_w + (size_t)l * DI_ * DC_, conv_b + (size_t)l * DI_, xc16);
    gemm_f16_wmma<DI_><<<dim3(MTOK / 128, 1), 256, 0, stream>>>(
        xc16, w16_x + (size_t)l * 48 * DI_, dbc, nullptr, nullptr, nullptr,
        48, 0);
    mamba_scan<<<B_, DI_, 0, stream>>>(
        dbc, xc16, xz, dt_w + (size_t)l * DI_ * R_, dt_b + (size_t)l * DI_,
        A_log + (size_t)l * DI_ * DS_, D_par + (size_t)l * DI_, y16);
    gemm_f16_wmma<DI_><<<dim3(MTOK / 128, DM_ / 64), 256, 0, stream>>>(
        y16, w16_out + (size_t)l * DM_ * DI_, h, h, nullptr, res_s,
        DM_, 1);
  }

  // ---- output LN + attention pooling + head ----
  layer_norm<<<MTOK, 256, 0, stream>>>(h, oln_g, oln_b, hn16, hn);
  gemm_f16_wmma<DM_><<<dim3(MTOK / 128, H_ / 64), 256, 0, stream>>>(
      hn16, w16_a1, t1, nullptr, a_b1, nullptr, H_, 2);
  attn_score<<<MTOK, H_, 0, stream>>>(t1, a_w2, sbuf);
  softmax_ctx<<<B_, 256, 0, stream>>>(sbuf, hn, lengths, ctxb);
  head_kernel<<<1, B_ * NC_, 0, stream>>>(ctxb, head_w, head_b, out);
}
